// MoleculeEncoder_50199577756097
// MI455X (gfx1250) — matrix operand order repaired and checked by reading, not compiled, not run on hardware
//
#include <hip/hip_runtime.h>
#include <hip/hip_bf16.h>
#include <math.h>

// ---------------------------------------------------------------- constants
typedef unsigned short u16;
typedef __attribute__((ext_vector_type(16))) __bf16 v16bf;
typedef __attribute__((ext_vector_type(8)))  float  v8f;

#define B_  64
#define S_  128
#define D_  1024
#define H_  16
#define L_  8
#define F_  3072
#define DH_ 64
#define M_  (B_*S_)          // 8192 token rows
#define NEGBIG (-1e9f)

// ---------------------------------------------------------------- helpers
__device__ __forceinline__ u16 f2bf(float f) {            // round-to-nearest-even
  unsigned u = __float_as_uint(f);
  u += 0x7FFFu + ((u >> 16) & 1u);
  return (u16)(u >> 16);
}

// ISA 10.2: generic pointer to LDS has the LDS byte address in its low 32 bits.
__device__ __forceinline__ unsigned lds_lo32(const void* p) {
  return (unsigned)(size_t)p;
}

#define ASYNC_LDS_B128(ldsaddr, gaddr)                                        \
  asm volatile("global_load_async_to_lds_b128 %0, %1, off"                    \
               :: "v"(ldsaddr), "v"(gaddr) : "memory")
#define WAIT_ASYNC0() asm volatile("s_wait_asynccnt 0x0" ::: "memory")

struct FragBF { union { v16bf v; u16 u[16]; }; };

// A-matrix 16x32 bf16 fragment (contiguous k = 16*(lane/16) + e, matching the B loaders)
__device__ __forceinline__ void load_fragA(FragBF& f, const u16* __restrict__ base,
                                           int pitch, int lane) {
  const int m  = lane & 15;
  const int hi = (lane >> 4) & 1;
  #pragma unroll
  for (int j = 0; j < 8; ++j) {
    const int kb = hi ? 16 : 0;
    const int k  = kb + j * 2;
    f.u[2*j]   = base[(size_t)m * pitch + k];
    f.u[2*j+1] = base[(size_t)m * pitch + k + 1];
  }
}

// B-matrix 32x16 fragment, source n-major (addr = n*pitch + k)
__device__ __forceinline__ void load_fragB_nmajor(FragBF& f, const u16* __restrict__ base,
                                                  int pitch, int lane) {
  const int n  = lane & 15;
  const int kb = (lane & 16) ? 16 : 0;
  #pragma unroll
  for (int e = 0; e < 16; ++e) f.u[e] = base[(size_t)n * pitch + kb + e];
}

// B-matrix 32x16 fragment, source k-major (addr = k*pitch + n)
__device__ __forceinline__ void load_fragB_kmajor(FragBF& f, const u16* __restrict__ base,
                                                  int pitch, int lane) {
  const int n  = lane & 15;
  const int kb = (lane & 16) ? 16 : 0;
  #pragma unroll
  for (int e = 0; e < 16; ++e) f.u[e] = base[(size_t)(kb + e) * pitch + n];
}

// ---------------------------------------------------------------- fp32 -> bf16
__global__ __launch_bounds__(256)
void f2bf_kernel(const float* __restrict__ in, u16* __restrict__ out, int n) {
  int i = blockIdx.x * 256 + threadIdx.x;
  if (i < n) out[i] = f2bf(in[i]);
}

// ---------------------------------------------------------------- embedding
__global__ __launch_bounds__(256)
void embed_kernel(const int* __restrict__ atoms, const int* __restrict__ neigh,
                  const float* __restrict__ aemb, const float* __restrict__ nemb,
                  float* __restrict__ xf, u16* __restrict__ xb) {
  const int bs = blockIdx.x;
  const int d  = threadIdx.x * 4;
  const int a  = atoms[bs];
  const int nn = neigh[bs];
  float4 av = a  ? *(const float4*)(aemb + (size_t)a  * D_ + d) : make_float4(0.f,0.f,0.f,0.f);
  float4 nv = nn ? *(const float4*)(nemb + (size_t)nn * D_ + d) : make_float4(0.f,0.f,0.f,0.f);
  float o[4] = { av.x+nv.x, av.y+nv.y, av.z+nv.z, av.w+nv.w };
  #pragma unroll
  for (int i = 0; i < 4; ++i) {
    xf[(size_t)bs * D_ + d + i] = o[i];
    xb[(size_t)bs * D_ + d + i] = f2bf(o[i]);
  }
}

// ---------------------------------------------------------------- GEMM (bf16 WMMA)
// C[M,N] = A[M,K] @ B[K,N] (+bias)(+resid)(gelu?)
// Block tile 128x128, 8 waves, wave tile 32x64 (8 WMMAs / K-step / wave).
// Double-buffered LDS pipeline: tile t+1's A is fetched with
// GLOBAL_LOAD_ASYNC_TO_LDS_B128 (ASYNCcnt) and B prefetched into VGPRs right
// after the barrier; the 8 WMMAs of tile t run while those are in flight;
// the B registers are drained into the next LDS buffer after the math.
#define TM 128
#define TN 128
#define TK 32
#define PA (TK + 8)          // LDS row pitch: 40 ushorts = 80 B (16B-aligned)

__global__ __launch_bounds__(256)
void gemm_bf16(const u16* __restrict__ A, const u16* __restrict__ Bm,
               const float* __restrict__ bias, const float* __restrict__ resid,
               float* __restrict__ outF, u16* __restrict__ outB,
               int N, int K, int act) {
  __shared__ u16 lA[2][TM][PA];     // row-major  [m][k]
  __shared__ u16 lB[2][TN][PA];     // transposed [n][k]

  const int t    = threadIdx.x;
  const int lane = t & 31;
  const int w    = t >> 5;
  const int bm   = blockIdx.y * TM;
  const int bn   = blockIdx.x * TN;
  const int wm   = w >> 1;              // 0..3  : 32-row strip
  const int wn   = w & 1;               // 0..1  : 64-col strip

  v8f zero = {};
  v8f acc[8];
  #pragma unroll
  for (int i = 0; i < 8; ++i) acc[i] = zero;

  // staging slots: A 128x32 (two rows/thread), B 32x128 (two k-rows/thread)
  const int ar = t >> 2,  ac = (t & 3)  * 8;   // A rows ar, ar+64
  const int br = t >> 4,  bc = (t & 15) * 8;   // B k-rows br, br+16

  const unsigned ldsA0_0 = lds_lo32(&lA[0][ar][ac]);
  const unsigned ldsA1_0 = lds_lo32(&lA[0][ar + 64][ac]);
  const unsigned ldsA0_1 = lds_lo32(&lA[1][ar][ac]);
  const unsigned ldsA1_1 = lds_lo32(&lA[1][ar + 64][ac]);
  const u16* gA0 = A + (size_t)(bm + ar)      * K + ac;
  const u16* gA1 = A + (size_t)(bm + ar + 64) * K + ac;
  const u16* gB0 = Bm + (size_t)br        * N + bn + bc;
  const u16* gB1 = Bm + (size_t)(br + 16) * N + bn + bc;

  // ---- prologue: stage tile 0 into buffer 0
  ASYNC_LDS_B128(ldsA0_0, gA0);
  ASYNC_LDS_B128(ldsA1_0, gA1);
  {
    union { uint4 q; u16 s[8]; } u0, u1;
    u0.q = *(const uint4*)(gB0);
    u1.q = *(const uint4*)(gB1);
    #pragma unroll
    for (int i = 0; i < 8; ++i) { lB[0][bc + i][br] = u0.s[i]; lB[0][bc + i][br + 16] = u1.s[i]; }
  }

  const int NT = K / TK;
  for (int it = 0; it < NT; ++it) {
    const int cur = it & 1;
    WAIT_ASYNC0();                     // this wave's async A stores for tile it
    __syncthreads();                   // all waves: tile-it B stores visible, buf cur^1 free

    // ---- prefetch tile it+1 (A via async->LDS, B into VGPRs)
    union { uint4 q; u16 s[8]; } nb0, nb1;
    const bool have = (it + 1 < NT);
    if (have) {
      const size_t k1 = (size_t)(it + 1) * TK;
      ASYNC_LDS_B128(cur ? ldsA0_0 : ldsA0_1, gA0 + k1);
      ASYNC_LDS_B128(cur ? ldsA1_0 : ldsA1_1, gA1 + k1);
      nb0.q = *(const uint4*)(gB0 + k1 * N);
      nb1.q = *(const uint4*)(gB1 + k1 * N);
    }

    // ---- compute tile it from buffer cur
    const u16 (*cA)[PA] = lA[cur];
    const u16 (*cB)[PA] = lB[cur];
    FragBF fa0, fa1, fb0, fb1, fb2, fb3;
    load_fragA(fa0, &cA[wm * 32][0],      PA, lane);
    load_fragA(fa1, &cA[wm * 32 + 16][0], PA, lane);
    load_fragB_nmajor(fb0, &cB[wn * 64][0],      PA, lane);
    load_fragB_nmajor(fb1, &cB[wn * 64 + 16][0], PA, lane);
    load_fragB_nmajor(fb2, &cB[wn * 64 + 32][0], PA, lane);
    load_fragB_nmajor(fb3, &cB[wn * 64 + 48][0], PA, lane);

    acc[0] = __builtin_amdgcn_wmma_f32_16x16x32_bf16(false, fa0.v, false, fb0.v, (short)0, acc[0], false, false);
    acc[1] = __builtin_amdgcn_wmma_f32_16x16x32_bf16(false, fa0.v, false, fb1.v, (short)0, acc[1], false, false);
    acc[2] = __builtin_amdgcn_wmma_f32_16x16x32_bf16(false, fa0.v, false, fb2.v, (short)0, acc[2], false, false);
    acc[3] = __builtin_amdgcn_wmma_f32_16x16x32_bf16(false, fa0.v, false, fb3.v, (short)0, acc[3], false, false);
    acc[4] = __builtin_amdgcn_wmma_f32_16x16x32_bf16(false, fa1.v, false, fb0.v, (short)0, acc[4], false, false);
    acc[5] = __builtin_amdgcn_wmma_f32_16x16x32_bf16(false, fa1.v, false, fb1.v, (short)0, acc[5], false, false);
    acc[6] = __builtin_amdgcn_wmma_f32_16x16x32_bf16(false, fa1.v, false, fb2.v, (short)0, acc[6], false, false);
    acc[7] = __builtin_amdgcn_wmma_f32_16x16x32_bf16(false, fa1.v, false, fb3.v, (short)0, acc[7], false, false);

    // ---- drain B prefetch into the next buffer (after the math)
    if (have) {
      u16 (*nB)[PA] = lB[cur ^ 1];
      #pragma unroll
      for (int i = 0; i < 8; ++i) { nB[bc + i][br] = nb0.s[i]; nB[bc + i][br + 16] = nb1.s[i]; }
    }
  }

  // epilogue: C/D layout n = lane&15, m = vgpr + (lane>=16 ? 8 : 0)
  const int nl   = lane & 15;
  const int madd = (lane & 16) ? 8 : 0;
  #pragma unroll
  for (int s = 0; s < 2; ++s) {
    #pragma unroll
    for (int nt = 0; nt < 4; ++nt) {
      const v8f a = acc[s * 4 + nt];
      const int n = bn + wn * 64 + nt * 16 + nl;
      const float bi = bias[n];
      #pragma unroll
      for (int e = 0; e < 8; ++e) {
        const int m = bm + wm * 32 + s * 16 + madd + e;
        float v = a[e] + bi;
        if (resid) v += resid[(size_t)m * N + n];
        if (act)   v = 0.5f * v * (1.0f + erff(v * 0.70710678118654752f)); // exact gelu
        if (outF) outF[(size_t)m * N + n] = v;
        if (outB) outB[(size_t)m * N + n] = f2bf(v);
      }
    }
  }
}

// ---------------------------------------------------------------- attention
// One wave per (b, h, 16-query tile). scores = QK^T*0.125 + bias(distances),
// softmax in fp32 LDS, then P@V — all matmuls on WMMA.
__global__ __launch_bounds__(32)
void attn_bf16(const u16* __restrict__ Q, const u16* __restrict__ Kmat,
               const u16* __restrict__ V, const int* __restrict__ dist,
               const float* __restrict__ demb, u16* __restrict__ O) {
  __shared__ float sc[16][S_];
  __shared__ u16   pb[16][S_];

  const int lane = threadIdx.x;
  const int blk  = blockIdx.x;
  const int qt   = blk & 7;
  const int h    = (blk >> 3) & (H_ - 1);
  const int b    = blk >> 7;

  const size_t rowQ = (size_t)(b * S_ + qt * 16);
  FragBF qf0, qf1;
  load_fragA(qf0, Q + rowQ * D_ + h * DH_,      D_, lane);
  load_fragA(qf1, Q + rowQ * D_ + h * DH_ + 32, D_, lane);

  const int nl   = lane & 15;
  const int madd = (lane & 16) ? 8 : 0;

  // ---- scores: 16 x 128
  for (int kt = 0; kt < S_ / 16; ++kt) {
    const u16* kb = Kmat + (size_t)(b * S_ + kt * 16) * D_ + h * DH_;
    FragBF kf0, kf1;                       // B[k=dh][n=key] : n-major, pitch D
    load_fragB_nmajor(kf0, kb,      D_, lane);
    load_fragB_nmajor(kf1, kb + 32, D_, lane);
    v8f acc = {};
    acc = __builtin_amdgcn_wmma_f32_16x16x32_bf16(false, qf0.v, false, kf0.v,
                                                  (short)0, acc, false, false);
    acc = __builtin_amdgcn_wmma_f32_16x16x32_bf16(false, qf1.v, false, kf1.v,
                                                  (short)0, acc, false, false);
    const int kk = kt * 16 + nl;
    #pragma unroll
    for (int e = 0; e < 8; ++e) {
      const int m    = madd + e;
      const int qrow = qt * 16 + m;
      const int dv   = dist[((size_t)b * S_ + qrow) * S_ + kk];
      const float bv = (dv == 0) ? NEGBIG : demb[dv * H_ + h];
      sc[m][kk] = acc[e] * 0.125f + bv;    // 1/sqrt(64)
    }
  }
  __syncthreads();

  // ---- softmax (fp32), one row per lane 0..15
  if (lane < 16) {
    float mx = -3.4e38f;
    for (int j = 0; j < S_; ++j) mx = fmaxf(mx, sc[lane][j]);
    float sum = 0.f;
    for (int j = 0; j < S_; ++j) { float ev = expf(sc[lane][j] - mx); sc[lane][j] = ev; sum += ev; }
    const float inv = 1.0f / sum;
    for (int j = 0; j < S_; ++j) pb[lane][j] = f2bf(sc[lane][j] * inv);
  }
  __syncthreads();

  // ---- P(16x128) @ V(128x64)
  for (int nt = 0; nt < 4; ++nt) {
    v8f acc = {};
    for (int ks = 0; ks < 4; ++ks) {
      FragBF pa, vb;
      load_fragA(pa, &pb[0][0] + ks * 32, S_, lane);
      load_fragB_kmajor(vb, V + ((size_t)(b * S_) + ks * 32) * D_ + h * DH_ + nt * 16,
                        D_, lane);
      acc = __builtin_amdgcn_wmma_f32_16x16x32_bf16(false, pa.v, false, vb.v,
                                                    (short)0, acc, false, false);
    }
    #pragma unroll
    for (int e = 0; e < 8; ++e)
      O[(rowQ + madd + e) * D_ + h * DH_ + nt * 16 + nl] = f2bf(acc[e]);
  }
}

// ---------------------------------------------------------------- LayerNorm
__global__ __launch_bounds__(256)
void ln_kernel(const float* __restrict__ y, const float* __restrict__ g,
               const float* __restrict__ bt, float* __restrict__ xf,
               u16* __restrict__ xb) {
  __shared__ float rs[256], rq[256];
  const int row = blockIdx.x, t = threadIdx.x;
  const float4 v = *(const float4*)(y + (size_t)row * D_ + t * 4);
  rs[t] = v.x + v.y + v.z + v.w;
  rq[t] = v.x*v.x + v.y*v.y + v.z*v.z + v.w*v.w;
  __syncthreads();
  for (int off = 128; off > 0; off >>= 1) {
    if (t < off) { rs[t] += rs[t + off]; rq[t] += rq[t + off]; }
    __syncthreads();
  }
  const float mean = rs[0] * (1.0f / D_);
  const float var  = rq[0] * (1.0f / D_) - mean * mean;
  const float inv  = rsqrtf(var + 1e-5f);
  const float vv[4] = { v.x, v.y, v.z, v.w };
  #pragma unroll
  for (int i = 0; i < 4; ++i) {
    const int d = t * 4 + i;
    const float xv = (vv[i] - mean) * inv * g[d] + bt[d];
    xf[(size_t)row * D_ + d] = xv;
    xb[(size_t)row * D_ + d] = f2bf(xv);
  }
}

// ---------------------------------------------------------------- launch
extern "C" void kernel_launch(void* const* d_in, const int* in_sizes, int n_in,
                              void* d_out, int out_size, void* d_ws, size_t ws_size,
                              hipStream_t stream) {
  (void)in_sizes; (void)n_in; (void)out_size; (void)ws_size;
  const int*   atoms = (const int*)  d_in[0];
  const int*   neigh = (const int*)  d_in[1];
  const int*   dist  = (const int*)  d_in[2];
  const float* aemb  = (const float*)d_in[3];
  const float* nemb  = (const float*)d_in[4];
  const float* demb  = (const float*)d_in[5];
  const float* wq = (const float*)d_in[6];  const float* bq = (const float*)d_in[7];
  const float* wk = (const float*)d_in[8];  const float* bk = (const float*)d_in[9];
  const float* wv = (const float*)d_in[10]; const float* bv = (const float*)d_in[11];
  const float* wo = (const float*)d_in[12]; const float* bo = (const float*)d_in[13];
  const float* g1 = (const float*)d_in[14]; const float* be1= (const float*)d_in[15];
  const float* w1 = (const float*)d_in[16]; const float* b1 = (const float*)d_in[17];
  const float* w2 = (const float*)d_in[18]; const float* b2 = (const float*)d_in[19];
  const float* g2 = (const float*)d_in[20]; const float* be2= (const float*)d_in[21];

  float* x = (float*)d_out;                 // fp32 activations; final LN2 leaves answer here
  char*  ws = (char*)d_ws;
  const size_t MB = 1024ull * 1024ull;      // total ws use: 180 MB
  u16*   wqB = (u16*)(ws +   0*MB);
  u16*   wkB = (u16*)(ws +   2*MB);
  u16*   wvB = (u16*)(ws +   4*MB);
  u16*   woB = (u16*)(ws +   6*MB);
  u16*   w1B = (u16*)(ws +   8*MB);
  u16*   w2B = (u16*)(ws +  14*MB);
  u16*   xB  = (u16*)(ws +  20*MB);
  u16*   qB  = (u16*)(ws +  36*MB);
  u16*   kB  = (u16*)(ws +  52*MB);
  u16*   vB  = (u16*)(ws +  68*MB);
  u16*   oB  = (u16*)(ws +  84*MB);
  float* y   = (float*)(ws + 100*MB);
  u16*   hB  = (u16*)(ws + 132*MB);

  // weights -> bf16 (shared across the 8 layers)
  f2bf_kernel<<<(D_*D_ + 255)/256, 256, 0, stream>>>(wq, wqB, D_*D_);
  f2bf_kernel<<<(D_*D_ + 255)/256, 256, 0, stream>>>(wk, wkB, D_*D_);
  f2bf_kernel<<<(D_*D_ + 255)/256, 256, 0, stream>>>(wv, wvB, D_*D_);
  f2bf_kernel<<<(D_*D_ + 255)/256, 256, 0, stream>>>(wo, woB, D_*D_);
  f2bf_kernel<<<(D_*F_ + 255)/256, 256, 0, stream>>>(w1, w1B, D_*F_);
  f2bf_kernel<<<(F_*D_ + 255)/256, 256, 0, stream>>>(w2, w2B, F_*D_);

  embed_kernel<<<M_, 256, 0, stream>>>(atoms, neigh, aemb, nemb, x, xB);

  const dim3 gD(D_/TN, M_/TM);   // 8 x 64
  const dim3 gF(F_/TN, M_/TM);   // 24 x 64
  for (int l = 0; l < L_; ++l) {
    gemm_bf16<<<gD, 256, 0, stream>>>(xB, wqB, bq, nullptr, nullptr, qB, D_, D_, 0);
    gemm_bf16<<<gD, 256, 0, stream>>>(xB, wkB, bk, nullptr, nullptr, kB, D_, D_, 0);
    gemm_bf16<<<gD, 256, 0, stream>>>(xB, wvB, bv, nullptr, nullptr, vB, D_, D_, 0);
    attn_bf16<<<B_*H_*(S_/16), 32, 0, stream>>>(qB, kB, vB, dist, demb, oB);
    gemm_bf16<<<gD, 256, 0, stream>>>(oB, woB, bo, x, y, nullptr, D_, D_, 0);
    ln_kernel<<<M_, 256, 0, stream>>>(y, g1, be1, x, xB);
    gemm_bf16<<<gF, 256, 0, stream>>>(xB, w1B, b1, nullptr, nullptr, hB, F_, D_, 1);
    gemm_bf16<<<gD, 256, 0, stream>>>(hB, w2B, b2, x, y, nullptr, D_, F_, 0);
    ln_kernel<<<M_, 256, 0, stream>>>(y, g2, be2, x, xB);
  }
}
